// CrossAttention_7395933683864
// MI455X (gfx1250) — matrix operand order repaired and checked by reading, not compiled, not run on hardware
//
#include <hip/hip_runtime.h>
#include <hip/hip_bf16.h>
#include <stdint.h>

// Problem constants (from reference): B=4, C=256, H=W=64 -> N=4096, CQK=32
#define NB  4
#define NC  256
#define ND  32
#define NN  4096

typedef __attribute__((ext_vector_type(16))) __bf16 v16bf;
typedef __attribute__((ext_vector_type(8)))  float  v8f;

union Frag {
  uint4 u[2];   // two 16B loads
  v16bf v;      // WMMA operand (8 VGPRs, 16 bf16 per lane)
};

__device__ __forceinline__ unsigned short f32_to_bf16(float x) {
  union { float f; unsigned u; } t; t.f = x;
  unsigned r = t.u + 0x7FFFu + ((t.u >> 16) & 1u);  // round-to-nearest-even
  return (unsigned short)(r >> 16);
}

// ---------------------------------------------------------------------------
// Kernel 1: fused 1x1-conv projections.
//   q[b][n][d]  (row-major over d, bf16)   <- Wq * f1 + bq
//   k[b][m][d]  (row-major over d, bf16)   <- Wk * f2 + bk
//   vt[b][c][m] (row-major over m, bf16)   <- (Wv * f2 + bv), stored transposed
// ---------------------------------------------------------------------------
__global__ __launch_bounds__(256)
void proj_kernel(const float* __restrict__ f1, const float* __restrict__ f2,
                 const float* __restrict__ Wq, const float* __restrict__ bq,
                 const float* __restrict__ Wk, const float* __restrict__ bk,
                 const float* __restrict__ Wv, const float* __restrict__ bv,
                 unsigned short* __restrict__ qb,
                 unsigned short* __restrict__ kb,
                 unsigned short* __restrict__ vtb)
{
  const int b  = blockIdx.y;
  const int p0 = blockIdx.x * 16;
  const int t  = threadIdx.x;

  __shared__ float x1s[NC * 16];
  __shared__ float x2s[NC * 16];

  const float* src1 = f1 + (size_t)b * NC * NN + p0;
  const float* src2 = f2 + (size_t)b * NC * NN + p0;

  for (int i = t; i < NC * 16; i += 256) {
    int c = i >> 4, px = i & 15;
    x1s[i] = src1[(size_t)c * NN + px];
    x2s[i] = src2[(size_t)c * NN + px];
  }
  __syncthreads();

  // 320 output rows (32 q + 32 k + 256 v) x 16 px = 5120 dots, 20 per thread.
  for (int j = 0; j < 20; ++j) {
    int o   = j * 256 + t;
    int px  = o & 15;
    int row = o >> 4;
    const float* w;
    const float* xs;
    float bias;
    if (row < 32)       { w = Wq + row * NC;        bias = bq[row];        xs = x1s; }
    else if (row < 64)  { w = Wk + (row - 32) * NC; bias = bk[row - 32];   xs = x2s; }
    else                { w = Wv + (row - 64) * NC; bias = bv[row - 64];   xs = x2s; }
    float acc = bias;
    #pragma unroll 8
    for (int c = 0; c < NC; ++c) acc = fmaf(w[c], xs[c * 16 + px], acc);
    unsigned short h = f32_to_bf16(acc);
    int n = p0 + px;
    if (row < 32)      qb [((size_t)b * NN + n) * ND + row]        = h;
    else if (row < 64) kb [((size_t)b * NN + n) * ND + (row - 32)] = h;
    else               vtb[((size_t)b * NC + (row - 64)) * NN + n] = h;
  }
}

// ---------------------------------------------------------------------------
// Kernel 2: exact softmax row maxes (pass 1 of split softmax).
// Wave = 16 queries, sweeps all 4096 keys with QK WMMAs keeping a LANE-LOCAL
// running max (no shuffles in the loop); one 4-step reduction at the end.
// ---------------------------------------------------------------------------
__global__ __launch_bounds__(256)
void rowmax_kernel(const unsigned short* __restrict__ qb,
                   const unsigned short* __restrict__ kb,
                   float* __restrict__ rmx)
{
  const int b    = blockIdx.y;
  const int wave = threadIdx.x >> 5;
  const int lane = threadIdx.x & 31;
  const int half = lane >> 4;
  const int l16  = lane & 15;
  const int n0   = blockIdx.x * 128 + wave * 16;

  const unsigned short* qrow = qb + ((size_t)b * NN + (n0 + l16)) * ND;
  Frag aq;
  aq.u[0] = *(const uint4*)(qrow + 16 * half);
  aq.u[1] = *(const uint4*)(qrow + 16 * half + 8);

  float mloc[8];
  #pragma unroll
  for (int r = 0; r < 8; ++r) mloc[r] = -__builtin_inff();

  const unsigned short* kbase = kb + (size_t)b * NN * ND;

  for (int m0 = 0; m0 < NN; m0 += 64) {
    if (m0 + 64 < NN)
      __builtin_prefetch(kbase + (size_t)(m0 + 64 + l16) * ND, 0, 0);
    Frag bk[4];
    #pragma unroll
    for (int t = 0; t < 4; ++t) {
      const unsigned short* kr = kbase + (size_t)(m0 + t * 16 + l16) * ND;
      bk[t].u[0] = *(const uint4*)(kr + 16 * half);
      bk[t].u[1] = *(const uint4*)(kr + 16 * half + 8);
    }
    v8f S[4];
    #pragma unroll
    for (int t = 0; t < 4; ++t) {
      v8f z = (v8f){0.f,0.f,0.f,0.f,0.f,0.f,0.f,0.f};
      S[t] = __builtin_amdgcn_wmma_f32_16x16x32_bf16(false, aq.v, false, bk[t].v,
                                                     (short)0, z, false, false);
    }
    #pragma unroll
    for (int r = 0; r < 8; ++r) {
      float mx = fmaxf(fmaxf(S[0][r], S[1][r]), fmaxf(S[2][r], S[3][r]));
      mloc[r] = fmaxf(mloc[r], mx);
    }
  }

  // One reduction across the 16-lane half per row, then store (2 lanes/row).
  #pragma unroll
  for (int r = 0; r < 8; ++r) {
    float mx = mloc[r];
    #pragma unroll
    for (int off = 1; off < 16; off <<= 1) mx = fmaxf(mx, __shfl_xor(mx, off, 32));
    if (l16 == 0) rmx[(size_t)b * NN + n0 + r + 8 * half] = mx;
  }
}

// ---------------------------------------------------------------------------
// Kernel 3: flash cross-attention with precomputed row maxes (pass 2).
// Block: 256 threads = 8 wave32s, owns 64 queries of one batch.
// Wave w: qg=w>>1 (16 queries), cg=w&1 (128 channels, 8 C-tiles, 64 acc VGPRs).
// Hot loop per 64 keys: 4 QK WMMAs, branch-free exp (uniform precomputed
// shift), P->LDS relayout, 16 AV WMMAs. No shuffles, no rescale in the loop.
// ---------------------------------------------------------------------------
#define PSTRIDE 88   // P-tile row stride in bf16: 176B (16B aligned), 44 dwords

__global__ __launch_bounds__(256)
void attn_kernel(const unsigned short* __restrict__ qb,
                 const unsigned short* __restrict__ kb,
                 const unsigned short* __restrict__ vtb,
                 const float* __restrict__ rmx,
                 float* __restrict__ out)
{
  const int b    = blockIdx.y;
  const int n0   = blockIdx.x * 64;
  const int wave = threadIdx.x >> 5;
  const int lane = threadIdx.x & 31;
  const int qg   = wave >> 1;        // 0..3  -> queries n0+qg*16 ..
  const int cg   = wave & 1;         // 0..1  -> channels cg*128 .. +127
  const int half = lane >> 4;
  const int l16  = lane & 15;

  __shared__ __align__(16) unsigned short pbuf[8][16][PSTRIDE];

  const unsigned short* qrow = qb + ((size_t)b * NN + (n0 + qg * 16 + l16)) * ND;
  Frag aq;
  aq.u[0] = *(const uint4*)(qrow + 16 * half);
  aq.u[1] = *(const uint4*)(qrow + 16 * half + 8);

  // Row maxes for this lane's 8 rows (rows 8h..8h+7): two 16B broadcast loads.
  const float* mrp = rmx + (size_t)b * NN + n0 + qg * 16 + 8 * half;
  float4 mA = *(const float4*)mrp;
  float4 mB = *(const float4*)(mrp + 4);
  float mr[8] = {mA.x, mA.y, mA.z, mA.w, mB.x, mB.y, mB.z, mB.w};

  v8f O[8];
  #pragma unroll
  for (int t = 0; t < 8; ++t) O[t] = (v8f){0.f,0.f,0.f,0.f,0.f,0.f,0.f,0.f};
  float lsum[8];
  #pragma unroll
  for (int r = 0; r < 8; ++r) lsum[r] = 0.f;

  const unsigned short* kbase = kb + (size_t)b * NN * ND;
  const unsigned short* vbase =
      vtb + ((size_t)b * NC + cg * 128 + l16) * (size_t)NN;

  for (int m0 = 0; m0 < NN; m0 += 64) {
    if (m0 + 64 < NN)
      __builtin_prefetch(kbase + (size_t)(m0 + 64 + l16) * ND, 0, 0);

    Frag bk[4];
    #pragma unroll
    for (int t = 0; t < 4; ++t) {
      const unsigned short* kr = kbase + (size_t)(m0 + t * 16 + l16) * ND;
      bk[t].u[0] = *(const uint4*)(kr + 16 * half);
      bk[t].u[1] = *(const uint4*)(kr + 16 * half + 8);
    }
    v8f S[4];
    #pragma unroll
    for (int t = 0; t < 4; ++t) {
      v8f z = (v8f){0.f,0.f,0.f,0.f,0.f,0.f,0.f,0.f};
      S[t] = __builtin_amdgcn_wmma_f32_16x16x32_bf16(false, aq.v, false, bk[t].v,
                                                     (short)0, z, false, false);
    }

    // Branch-free softmax numerator with uniform precomputed shift.
    #pragma unroll
    for (int r = 0; r < 8; ++r) {
      float m = mr[r];
      float p0 = __expf(S[0][r] - m);
      float p1 = __expf(S[1][r] - m);
      float p2 = __expf(S[2][r] - m);
      float p3 = __expf(S[3][r] - m);
      lsum[r] += (p0 + p1) + (p2 + p3);   // lane-partial, reduced at the end
      int row = r + 8 * half;
      pbuf[wave][row][ 0 + l16] = f32_to_bf16(p0);
      pbuf[wave][row][16 + l16] = f32_to_bf16(p1);
      pbuf[wave][row][32 + l16] = f32_to_bf16(p2);
      pbuf[wave][row][48 + l16] = f32_to_bf16(p3);
    }

    // Re-load P as two A fragments (16 q x 32 keys each) from this wave's tile.
    const unsigned short* prow = &pbuf[wave][l16][0];
    Frag ap0, ap1;
    ap0.u[0] = *(const uint4*)(prow + 16 * half);
    ap0.u[1] = *(const uint4*)(prow + 16 * half + 8);
    ap1.u[0] = *(const uint4*)(prow + 32 + 16 * half);
    ap1.u[1] = *(const uint4*)(prow + 32 + 16 * half + 8);

    // O += P * V^T over 8 channel subtiles x 2 key halves.
    #pragma unroll
    for (int t = 0; t < 8; ++t) {
      const unsigned short* vr = vbase + (size_t)t * (16 * NN) + m0;
      Frag bv0, bv1;
      bv0.u[0] = *(const uint4*)(vr + 16 * half);
      bv0.u[1] = *(const uint4*)(vr + 16 * half + 8);
      bv1.u[0] = *(const uint4*)(vr + 32 + 16 * half);
      bv1.u[1] = *(const uint4*)(vr + 32 + 16 * half + 8);
      O[t] = __builtin_amdgcn_wmma_f32_16x16x32_bf16(false, ap0.v, false, bv0.v,
                                                     (short)0, O[t], false, false);
      O[t] = __builtin_amdgcn_wmma_f32_16x16x32_bf16(false, ap1.v, false, bv1.v,
                                                     (short)0, O[t], false, false);
    }
  }

  // One softmax-sum reduction, normalize, store out[b][c][n] f32.
  #pragma unroll
  for (int r = 0; r < 8; ++r) {
    float ps = lsum[r];
    #pragma unroll
    for (int off = 1; off < 16; off <<= 1) ps += __shfl_xor(ps, off, 32);
    float inv = 1.0f / ps;
    int n = n0 + qg * 16 + r + 8 * half;
    #pragma unroll
    for (int t = 0; t < 8; ++t) {
      int c = cg * 128 + t * 16 + l16;
      out[((size_t)b * NC + c) * (size_t)NN + n] = O[t][r] * inv;
    }
  }
}

// ---------------------------------------------------------------------------
// Launch. Workspace layout:
//   qb : NB*NN*ND bf16 (1 MB)
//   kb : NB*NN*ND bf16 (1 MB)
//   vtb: NB*NC*NN bf16 (8 MB)
//   rmx: NB*NN f32     (64 KB)        total ~10.6 MB of d_ws
// ---------------------------------------------------------------------------
extern "C" void kernel_launch(void* const* d_in, const int* in_sizes, int n_in,
                              void* d_out, int out_size, void* d_ws, size_t ws_size,
                              hipStream_t stream) {
  const float* f1 = (const float*)d_in[0];
  const float* f2 = (const float*)d_in[1];
  const float* Wq = (const float*)d_in[2];
  const float* bq = (const float*)d_in[3];
  const float* Wk = (const float*)d_in[4];
  const float* bk = (const float*)d_in[5];
  const float* Wv = (const float*)d_in[6];
  const float* bv = (const float*)d_in[7];

  unsigned short* qb  = (unsigned short*)d_ws;
  unsigned short* kb  = qb + (size_t)NB * NN * ND;
  unsigned short* vtb = kb + (size_t)NB * NN * ND;
  float*          rmx = (float*)(vtb + (size_t)NB * NC * NN);

  proj_kernel<<<dim3(NN / 16, NB), 256, 0, stream>>>(f1, f2, Wq, bq, Wk, bk, Wv, bv,
                                                     qb, kb, vtb);
  rowmax_kernel<<<dim3(NN / 128, NB), 256, 0, stream>>>(qb, kb, rmx);
  attn_kernel<<<dim3(NN / 64, NB), 256, 0, stream>>>(qb, kb, vtb, rmx, (float*)d_out);
}
